// BCQLinear_88510686036218
// MI455X (gfx1250) — compile-verified, matrix operand order checked by reading, not run on hardware
//
#include <hip/hip_runtime.h>
#include <hip/hip_bf16.h>

// ---------------- problem constants (from reference) ----------------
#define IN_F   4096
#define OUT_F  4096
#define WBITS  3
#define GS     128
#define OFI    128
#define NG     (IN_F / GS)              // 32
#define NB     (OUT_F / OFI)            // 32
#define WPB    (WBITS * GS * OFI / 32)  // 1536 words per (g, n) block
#define MTOK   128                      // tokens (B*T = 1*128)

#define GSPLIT 4                        // K-split over groups
#define GPW    (NG / GSPLIT)            // 8 groups per workgroup
#define PSLICE ((size_t)MTOK * OUT_F)

typedef __attribute__((ext_vector_type(16))) _Float16 v16h;
typedef __attribute__((ext_vector_type(8)))  float    v8f;

union FragU  { v16h h; unsigned u[8]; };
union HPack  { _Float16 h[2]; unsigned u; };

__device__ __forceinline__ unsigned lds_addr(const void* p) {
    // __shared__ generic pointer: low 32 bits are the LDS byte offset
    return (unsigned)(unsigned long long)p;
}

__device__ __forceinline__ void async_load_lds_b128(unsigned lds, const void* gaddr) {
    unsigned long long ga = (unsigned long long)gaddr;
    asm volatile("global_load_async_to_lds_b128 %0, %1, off"
                 :: "v"(lds), "v"(ga) : "memory");
}

__device__ __forceinline__ void wait_asynccnt0() {
    asm volatile("s_wait_asynccnt 0x0" ::: "memory");
}

// ---------------- kernel 1: x gather (in_reorder) + f32 -> f16 ----------------
__global__ __launch_bounds__(256)
void bcq_permute_x(const float* __restrict__ x,
                   const int* __restrict__ in_reorder,
                   _Float16* __restrict__ xr)
{
    int idx = blockIdx.x * 256 + threadIdx.x;           // 0 .. 128*4096-1
    int t = idx >> 12;
    int i = idx & (IN_F - 1);
    int src = in_reorder[i];
    xr[(size_t)t * IN_F + i] = (_Float16)x[(size_t)t * IN_F + src];
}

// ---------------- kernel 2: fused dequant (LDS) + WMMA GEMM ----------------
// grid.x = NB (out-block), grid.y = GSPLIT (K split). 256 threads = 8 waves.
// Each WG: C = 128 tokens x 128 out cols, accumulated over 8 groups.
// qwords are DMA'd global->LDS with GLOBAL_LOAD_ASYNC_TO_LDS_B128, double
// buffered so the DMA for group g+1 overlaps the WMMA phase of group g.
__global__ __launch_bounds__(256)
void bcq_gemm(const int* __restrict__ qweight,
              const float* __restrict__ alpha,
              const float* __restrict__ beta,
              const int* __restrict__ offset,
              const _Float16* __restrict__ xr,
              float* __restrict__ partial)
{
    __shared__ _Float16 sW[OFI * GS];        // transposed: sW[n*128 + k], 32 KB
    __shared__ unsigned sWords[2][WPB];      // double-buffered packed words, 12 KB
    __shared__ float    sA2[OFI];            // 2*alpha
    __shared__ float    sBB[OFI];            // beta - 7*alpha

    const int tid    = threadIdx.x;
    const int lane   = tid & 31;
    const int wv     = tid >> 5;             // wave id 0..7
    const int nb     = blockIdx.x;           // out-block
    const int gbase  = blockIdx.y * GPW;     // first group for this WG
    const int mstrip = wv * 16;              // this wave's token rows

    const int lm  = lane & 15;               // fragment row/col within tile
    const int khi = (lane & 16) ? 8 : 0;     // lane-half K offset (A/B layout)
    const int chi = (lane & 16) ? 8 : 0;     // lane-half M offset (C layout)

    v8f acc[8];
    const v8f vzero = {0.f,0.f,0.f,0.f,0.f,0.f,0.f,0.f};
    #pragma unroll
    for (int nt = 0; nt < 8; ++nt) acc[nt] = vzero;

    // ---- prologue: async DMA of group gbase's 1536 words into buffer 0 ----
    {
        const int base = offset[gbase * NB + nb];
        const int* src = qweight + base;
        async_load_lds_b128(lds_addr(&sWords[0][tid * 4]), src + tid * 4);       // 1024 words
        if (tid < 128)
            async_load_lds_b128(lds_addr(&sWords[0][1024 + tid * 4]),
                                src + 1024 + tid * 4);                           // 512 words
    }
    // prefetch group gbase's alpha/beta into registers
    float rA = 0.f, rB = 0.f;
    if (tid < OFI) {
        rA = alpha[(size_t)gbase * OUT_F + nb * OFI + tid];
        rB = beta [(size_t)gbase * OUT_F + nb * OFI + tid];
    }

    int buf = 0;
    for (int g = 0; g < GPW; ++g) {
        const int gg = gbase + g;

        // words[buf] DMA complete, publish alpha/beta, sync (also closes the
        // previous iteration's WMMA reads of sW before dequant overwrites it)
        wait_asynccnt0();
        if (tid < OFI) { sA2[tid] = 2.0f * rA; sBB[tid] = rB - 7.0f * rA; }
        __syncthreads();

        // ---- dequant: 3 bit-planes -> f16 pairs, transposed into sW[n][k] ----
        // thread -> (row pair i0..i0+1, 32-col chunk oc); one b32 store per col
        {
            const int ip  = tid >> 2;            // 0..63  -> i0 = 2*ip
            const int oc  = (tid & 3) << 5;      // 0,32,64,96
            const int i0  = ip << 1;
            const int wA  = i0 * 4 + (tid & 3);  // word of row i0
            const int wB  = wA + 4;              // word of row i0+1
            const unsigned a0 = sWords[buf][wA],        a1 = sWords[buf][wB];
            const unsigned b0 = sWords[buf][512 + wA],  b1 = sWords[buf][512 + wB];
            const unsigned c0 = sWords[buf][1024 + wA], c1 = sWords[buf][1024 + wB];
            #pragma unroll
            for (int j = 0; j < 32; ++j) {
                const int o = oc + j;
                const int m0 = (int)(((a0 >> j) & 1u) | (((b0 >> j) & 1u) << 1) |
                                     (((c0 >> j) & 1u) << 2));
                const int m1 = (int)(((a1 >> j) & 1u) | (((b1 >> j) & 1u) << 1) |
                                     (((c1 >> j) & 1u) << 2));
                const float a2 = sA2[o], bb = sBB[o];
                HPack pk;
                pk.h[0] = (_Float16)fmaf((float)m0, a2, bb);
                pk.h[1] = (_Float16)fmaf((float)m1, a2, bb);
                *(unsigned*)&sW[o * GS + i0] = pk.u;     // aligned b32 store
            }
        }
        __syncthreads();

        // ---- kick off next group's DMA + scale prefetch (overlaps WMMA) ----
        if (g + 1 < GPW) {
            const int nbase = offset[(gg + 1) * NB + nb];
            const int* nsrc = qweight + nbase;
            const int nbuf  = buf ^ 1;
            async_load_lds_b128(lds_addr(&sWords[nbuf][tid * 4]), nsrc + tid * 4);
            if (tid < 128)
                async_load_lds_b128(lds_addr(&sWords[nbuf][1024 + tid * 4]),
                                    nsrc + 1024 + tid * 4);
            if (tid < OFI) {
                rA = alpha[(size_t)(gg + 1) * OUT_F + nb * OFI + tid];
                rB = beta [(size_t)(gg + 1) * OUT_F + nb * OFI + tid];
            }
        }

        // ---- WMMA: A (16x32 f16 from xr) x B (32x16 f16 from sW) ----
        const _Float16* xrow = xr + (size_t)(mstrip + lm) * IN_F + gg * GS;
        #pragma unroll
        for (int k0 = 0; k0 < GS; k0 += 32) {
            FragU a;
            #pragma unroll
            for (int v = 0; v < 8; ++v) {
                const int k = k0 + ((v & 4) ? 16 : 0) + khi + 2 * (v & 3);
                a.u[v] = *(const unsigned*)(xrow + k);   // packed f16 pair
            }
            #pragma unroll
            for (int nt = 0; nt < 8; ++nt) {
                FragU b;
                const int n = nt * 16 + lm;
                #pragma unroll
                for (int v = 0; v < 8; ++v) {
                    const int k = k0 + ((v & 4) ? 16 : 0) + khi + 2 * (v & 3);
                    b.u[v] = *(const unsigned*)&sW[n * GS + k];
                }
                acc[nt] = __builtin_amdgcn_wmma_f32_16x16x32_f16(
                    false, a.h, false, b.h, (short)0, acc[nt], false, false);
            }
        }
        buf ^= 1;
    }

    // ---- write partial C (deterministic, no atomics) ----
    float* pout = partial + (size_t)blockIdx.y * PSLICE;
    #pragma unroll
    for (int nt = 0; nt < 8; ++nt) {
        const int n = nb * OFI + nt * 16 + lm;
        #pragma unroll
        for (int v = 0; v < 8; ++v) {
            const int m = mstrip + v + chi;
            pout[(size_t)m * OUT_F + n] = acc[nt][v];
        }
    }
}

// ---------------- kernel 3: reduce K-split partials + out_reorder ----------------
__global__ __launch_bounds__(256)
void bcq_reduce_permute(const float* __restrict__ partial,
                        const int* __restrict__ out_reorder,
                        float* __restrict__ out)
{
    int idx = blockIdx.x * 256 + threadIdx.x;           // 0 .. 128*4096-1
    int t = idx >> 12;
    int j = idx & (OUT_F - 1);
    int src = out_reorder[j];
    size_t base = (size_t)t * OUT_F + src;
    float s = partial[base] + partial[PSLICE + base]
            + partial[2 * PSLICE + base] + partial[3 * PSLICE + base];
    out[(size_t)t * OUT_F + j] = s;
}

// ---------------- launcher ----------------
extern "C" void kernel_launch(void* const* d_in, const int* in_sizes, int n_in,
                              void* d_out, int out_size, void* d_ws, size_t ws_size,
                              hipStream_t stream)
{
    const float* x           = (const float*)d_in[0];
    const int*   qweight     = (const int*)  d_in[1];
    const float* alpha       = (const float*)d_in[2];
    const float* beta        = (const float*)d_in[3];
    // d_in[4] = block_bitwidth (uniform == 3, unused)
    const int*   offset      = (const int*)  d_in[5];
    const int*   in_reorder  = (const int*)  d_in[6];
    const int*   out_reorder = (const int*)  d_in[7];
    float* out = (float*)d_out;

    // workspace layout: [ xr f16 (1 MB) | partial f32 x GSPLIT (8 MB) ]
    _Float16* xr      = (_Float16*)d_ws;
    float*    partial = (float*)((char*)d_ws + (size_t)MTOK * IN_F * sizeof(_Float16));

    bcq_permute_x<<<(MTOK * IN_F) / 256, 256, 0, stream>>>(x, in_reorder, xr);

    dim3 grid(NB, GSPLIT);
    bcq_gemm<<<grid, 256, 0, stream>>>(qweight, alpha, beta, offset, xr, partial);

    bcq_reduce_permute<<<(MTOK * OUT_F) / 256, 256, 0, stream>>>(partial, out_reorder, out);
}
